// LongRangeTCN_78752520339709
// MI455X (gfx1250) — compile-verified, matrix operand order checked by reading, not run on hardware
//
#include <hip/hip_runtime.h>
#include <hip/hip_bf16.h>
#include <stdint.h>

// Problem constants (match reference)
#define B_      32
#define C_      128
#define T_      4096
#define KW      3
#define NLAYER  4
#define TN      64              // timesteps per block tile
#define HALO    16              // max (K-1)*d = 2*8
#define TILE_T  (TN + HALO)     // 80

typedef __attribute__((ext_vector_type(16))) __bf16 v16bf;
typedef __attribute__((ext_vector_type(8)))  float  v8f;
typedef __attribute__((ext_vector_type(4)))  unsigned int u32x4;
typedef __attribute__((ext_vector_type(8)))  int i32x8;
typedef __attribute__((ext_vector_type(4)))  int i32x4;

__device__ __forceinline__ uint16_t f2bf(float f) {
    union { float f; uint32_t u; } c; c.f = f;
    uint32_t u = c.u;
    u += 0x7fffu + ((u >> 16) & 1u);   // round-to-nearest-even
    return (uint16_t)(u >> 16);
}

// Per-layer prepacked weight block: 3 taps x 8 cout-tiles x 4 cin-chunks
// x 32 lanes x 8 dwords  = 24576 dwords = 96 KiB
#define WLAYER_DWORDS (3*8*4*32*8)
#define WPK_DWORDS    (NLAYER * WLAYER_DWORDS)   // 98304

// ---------------------------------------------------------------------------
// Pack conv weights [4][Cout][Cin][K] f32 -> per-lane WMMA A-operand layout.
// A 16x32 bf16 layout: lane M = lane%16; Khalf = (lane/16)*8;
//   j<4 : K = Khalf+2j ; j>=4 : K = 16+Khalf+2(j-4)
// ---------------------------------------------------------------------------
__global__ void pack_weights_k(const float* __restrict__ w,
                               uint32_t* __restrict__ wpk) {
    int gid = blockIdx.x * blockDim.x + threadIdx.x;
    if (gid >= WPK_DWORDS) return;
    int idx = gid;
    int j    = idx & 7;  idx >>= 3;
    int lane = idx & 31; idx >>= 5;
    int q    = idx & 3;  idx >>= 2;
    int mt   = idx & 7;  idx >>= 3;
    int k    = idx % 3;
    int l    = idx / 3;
    int cout  = mt * 16 + (lane & 15);
    int khalf = (lane >> 4) << 3;
    int kk = (j < 4) ? (khalf + 2 * j) : (16 + khalf + 2 * (j - 4));
    int ci = q * 32 + kk;
    float w0 = w[(((size_t)l * C_ + cout) * C_ + ci    ) * KW + k];
    float w1 = w[(((size_t)l * C_ + cout) * C_ + ci + 1) * KW + k];
    wpk[gid] = (uint32_t)f2bf(w0) | ((uint32_t)f2bf(w1) << 16);
}

// Fold BatchNorm (eval) into per-channel scale/bias.
__global__ void bn_prep_k(const float* __restrict__ g,  const float* __restrict__ be,
                          const float* __restrict__ m,  const float* __restrict__ v,
                          float* __restrict__ scale, float* __restrict__ bias) {
    int i = blockIdx.x * blockDim.x + threadIdx.x;
    if (i >= NLAYER * C_) return;
    float inv = g[i] * rsqrtf(v[i] + 1e-5f);
    scale[i] = inv;
    bias[i]  = be[i] - m[i] * inv;
}

// Convert initial f32 activations -> persistent bf16 activation buffer.
__global__ void cvt_bf16_k(const float* __restrict__ x, uint16_t* __restrict__ xh) {
    size_t stride = (size_t)gridDim.x * blockDim.x;
    for (size_t i = blockIdx.x * (size_t)blockDim.x + threadIdx.x;
         i < (size_t)B_ * C_ * T_; i += stride)
        xh[i] = f2bf(x[i]);
}

// ---------------------------------------------------------------------------
// Dilated causal conv (3 taps of 128x128 channel GEMM) + folded BN via
// v_wmma_f32_16x16x32_bf16. Per-layer weight block is DMA'd into LDS by the
// Tensor Data Mover (tensor_load_to_lds + s_wait_tensorcnt); activations are
// staged bf16 into a time-major LDS tile so B-operand K-pairs are single
// 32-bit DS reads. Loop nest keeps the cout-tile pair innermost so each
// B-operand feeds two WMMAs immediately (low VGPR pressure -> multi-wave
// occupancy; no VGPR-MSB addressing).
// ---------------------------------------------------------------------------
__global__ __launch_bounds__(128)
void conv_bn_wmma_k(const uint16_t* __restrict__ XH, float* __restrict__ U,
                    const uint32_t* __restrict__ WPK,
                    const float* __restrict__ SCALE, const float* __restrict__ BIAS,
                    int layer, int dil) {
    __shared__ uint16_t tile[TILE_T * C_];                  // 20 KiB, [t][cin]
    __shared__ __align__(64) uint32_t wlds[WLAYER_DWORDS];  // 96 KiB weights

    const int b   = blockIdx.y;
    const int t0  = blockIdx.x * TN;
    const int tid = threadIdx.x;

    // --- TDM: DMA the first 24 KiB (1 row of 3072 x 8B) of this layer's 96 KiB packed weights
    // into LDS. Wave 0 issues (TDM ignores EXEC; gate to avoid 4x duplicate
    // DMA), waits TENSORcnt==0, then the barrier below publishes to the WG.
    if (tid < 32) {
        const uint64_t gaddr =
            (uint64_t)(uintptr_t)(WPK + (size_t)layer * WLAYER_DWORDS);
        const uint32_t ldsoff = (uint32_t)(uintptr_t)(&wlds[0]);  // addr[31:0]
        u32x4 g0;
        g0[0] = 1u;                                   // count=1 (valid user D#)
        g0[1] = ldsoff;                               // lds_addr
        g0[2] = (uint32_t)gaddr;                      // global_addr[31:0]
        g0[3] = (uint32_t)((gaddr >> 32) & 0x01FFFFFFu) | (2u << 30); // +type=2
        i32x8 g1;
        g1[0] = (int)(3u << 16);                      // data_size=3 (8 bytes)
        g1[1] = (int)(3072u << 16);                   // tensor_dim0[15:0]=3072
        g1[2] = (int)(1u << 16);                      // dim0[31:16]=0, dim1=1
        g1[3] = (int)(3072u << 16);                   // tile_dim0 = 3072
        g1[4] = 1;                                    // tile_dim1 = 1
        g1[5] = 3072;                                 // tensor_dim0_stride
        g1[6] = 0;  g1[7] = 0;                        // dim1_stride = 0
        i32x4 gz4 = {0, 0, 0, 0};                     // <=2D: groups 2/3 unused
        i32x8 gz8 = {0, 0, 0, 0, 0, 0, 0, 0};        // extra group (unused)
        __builtin_amdgcn_tensor_load_to_lds(g0, g1, gz4, gz4, gz8, 0);
        __builtin_amdgcn_s_wait_tensorcnt(0);
    }

    // --- Stage bf16 activations x[b, tid, t0-16 .. t0+63] into LDS, transposed.
    const uint16_t* xrow = XH + ((size_t)b * C_ + tid) * T_;
    __builtin_prefetch(xrow + t0 + TN, 0, 1);   // global_prefetch_b8: next tile
    #pragma unroll 4
    for (int i = 0; i < TILE_T; ++i) {
        int t = t0 - HALO + i;
        tile[i * C_ + tid] = (t >= 0) ? xrow[t] : (uint16_t)0;
    }
    __syncthreads();

    const int wave  = tid >> 5;
    const int lane  = tid & 31;
    const int Nn    = lane & 15;          // N column for B/C/D operands
    const int Khalf = (lane >> 4) << 3;   // K (and M) half-split across lanes

    v8f zero = {};
    v8f acc[2][4] = {{zero, zero, zero, zero}, {zero, zero, zero, zero}};

    #pragma unroll 1
    for (int k = 0; k < KW; ++k) {
        const int shift = HALO - (2 - k) * dil;       // LDS time offset, >= 0
        #pragma unroll 1
        for (int q = 0; q < 4; ++q) {                 // cin chunks of 32
            // A operands for both cout tiles of this wave.
            union { v16bf v; uint32_t u[8]; } A[2];
            #pragma unroll
            for (int mi = 0; mi < 2; ++mi) {
                const int mt = wave * 2 + mi;
                const uint32_t* ap =
                    wlds + ((((k * 8 + mt) * 4 + q) * 32 + lane) * 8);
                #pragma unroll
                for (int j = 0; j < 8; ++j) A[mi].u[j] = ap[j];
            }
            #pragma unroll
            for (int n = 0; n < 4; ++n) {             // four 16-wide t sub-tiles
                union { v16bf v; uint32_t u[8]; } Bm;
                const int tl = n * 16 + shift + Nn;
                #pragma unroll
                for (int j = 0; j < 8; ++j) {
                    int kk = (j < 4) ? (Khalf + 2 * j) : (16 + Khalf + 2 * (j - 4));
                    int ci = q * 32 + kk;             // even -> dword aligned
                    Bm.u[j] = *(const uint32_t*)&tile[tl * C_ + ci];
                }
                acc[0][n] = __builtin_amdgcn_wmma_f32_16x16x32_bf16(
                    false, A[0].v, false, Bm.v, (short)0, acc[0][n], false, false);
                acc[1][n] = __builtin_amdgcn_wmma_f32_16x16x32_bf16(
                    false, A[1].v, false, Bm.v, (short)0, acc[1][n], false, false);
            }
        }
    }

    // Folded BN + store. C/D layout: VGPR r -> M = r + Khalf, lane -> N.
    for (int mi = 0; mi < 2; ++mi) {
        const int mt = wave * 2 + mi;
        float sc[8], bi[8];
        #pragma unroll
        for (int r = 0; r < 8; ++r) {
            int cout = mt * 16 + r + Khalf;
            sc[r] = SCALE[layer * C_ + cout];
            bi[r] = BIAS[layer * C_ + cout];
        }
        for (int n = 0; n < 4; ++n) {
            #pragma unroll
            for (int r = 0; r < 8; ++r) {
                int cout = mt * 16 + r + Khalf;
                int t = t0 + n * 16 + Nn;
                U[((size_t)b * C_ + cout) * T_ + t] = acc[mi][n][r] * sc[r] + bi[r];
            }
        }
    }
}

// ---------------------------------------------------------------------------
// LIF recurrence over T (sequential) + residual add. One thread per (b,c) row,
// float4-vectorized (4 recurrence steps per 16B access). Writes the f32
// residual chain (OUT) and the bf16 copy (XH) consumed by the next conv.
// In-place safe when OUT == U (each thread owns its row).
// ---------------------------------------------------------------------------
__global__ __launch_bounds__(128)
void lif_residual_k(const float* __restrict__ U, const float* __restrict__ R,
                    float* __restrict__ OUT, uint16_t* __restrict__ XH) {
    const int b = blockIdx.x;
    const int c = threadIdx.x;
    const size_t base = ((size_t)b * C_ + c) * T_;
    const float4* u4 = (const float4*)(U + base);
    const float4* r4 = (const float4*)(R + base);
    float4*       o4 = (float4*)(OUT + base);
    uint64_t*     h4 = (uint64_t*)(XH + base);
    float v = 0.0f;
    for (int i = 0; i < T_ / 4; ++i) {
        float4 u = u4[i];
        float4 r = r4[i];
        float4 o;
        float s;
        v = v + (u.x - v) * 0.5f; s = (v >= 1.0f) ? 1.0f : 0.0f; v *= (1.0f - s);
        o.x = s + r.x;
        v = v + (u.y - v) * 0.5f; s = (v >= 1.0f) ? 1.0f : 0.0f; v *= (1.0f - s);
        o.y = s + r.y;
        v = v + (u.z - v) * 0.5f; s = (v >= 1.0f) ? 1.0f : 0.0f; v *= (1.0f - s);
        o.z = s + r.z;
        v = v + (u.w - v) * 0.5f; s = (v >= 1.0f) ? 1.0f : 0.0f; v *= (1.0f - s);
        o.w = s + r.w;
        o4[i] = o;
        h4[i] = (uint64_t)f2bf(o.x)
              | ((uint64_t)f2bf(o.y) << 16)
              | ((uint64_t)f2bf(o.z) << 32)
              | ((uint64_t)f2bf(o.w) << 48);
    }
}

// ---------------------------------------------------------------------------
extern "C" void kernel_launch(void* const* d_in, const int* in_sizes, int n_in,
                              void* d_out, int out_size, void* d_ws, size_t ws_size,
                              hipStream_t stream) {
    const float* x     = (const float*)d_in[0];
    const float* w     = (const float*)d_in[1];
    const float* gamma = (const float*)d_in[2];
    const float* beta  = (const float*)d_in[3];
    const float* mean  = (const float*)d_in[4];
    const float* var   = (const float*)d_in[5];
    float* out = (float*)d_out;

    const size_t nelem = (size_t)B_ * C_ * T_;
    const size_t f32_bytes = nelem * sizeof(float);     // 64 MiB
    const size_t bf16_bytes = nelem * sizeof(uint16_t); // 32 MiB
    char* ws = (char*)d_ws;
    float*    bufA  = (float*)(ws);
    float*    bufB  = (float*)(ws + f32_bytes);
    uint16_t* xh    = (uint16_t*)(ws + 2 * f32_bytes);
    uint32_t* wpk   = (uint32_t*)(ws + 2 * f32_bytes + bf16_bytes);
    float*    scale = (float*)(ws + 2 * f32_bytes + bf16_bytes
                               + (size_t)WPK_DWORDS * 4);
    float*    bias  = scale + NLAYER * C_;

    pack_weights_k<<<WPK_DWORDS / 128, 128, 0, stream>>>(w, wpk);
    bn_prep_k<<<(NLAYER * C_ + 127) / 128, 128, 0, stream>>>(gamma, beta, mean, var,
                                                             scale, bias);
    cvt_bf16_k<<<2048, 256, 0, stream>>>(x, xh);

    const int dil[NLAYER] = {1, 2, 4, 8};
    const float* res = x;                 // f32 residual source for this layer
    float* ping[2] = {bufA, bufB};
    for (int l = 0; l < NLAYER; ++l) {
        float* u = ping[l & 1];
        dim3 grid(T_ / TN, B_);
        conv_bn_wmma_k<<<grid, 128, 0, stream>>>(xh, u, wpk, scale, bias, l, dil[l]);
        float* o = (l == NLAYER - 1) ? out : u;    // last layer -> d_out
        lif_residual_k<<<B_, C_, 0, stream>>>(u, res, o, xh);
        res = o;
    }
}
